// CapsNet_33509334843728
// MI455X (gfx1250) — matrix operand order repaired and checked by reading, not compiled, not run on hardware
//
#include <hip/hip_runtime.h>
#include <hip/hip_bf16.h>

typedef _Float16 f16;
typedef _Float16 v16h __attribute__((ext_vector_type(16)));
typedef _Float16 v8h  __attribute__((ext_vector_type(8)));
typedef float    v8f  __attribute__((ext_vector_type(8)));

#define NBATCH 180

// ---------------------------------------------------------------------------
// gfx1250 async global->LDS copy (ASYNCcnt-tracked), 16 bytes per lane.
// VDST operand is a VGPR holding the LDS byte address (low 32 bits of the
// generic pointer are the LDS offset per the flat-addressing rules).
// ---------------------------------------------------------------------------
__device__ __forceinline__ void async_ld16(void* lds_dst, const void* gsrc) {
    unsigned loff = (unsigned)(unsigned long long)lds_dst;
    asm volatile("global_load_async_to_lds_b128 %0, %1, off"
                 :: "v"(loff), "v"(gsrc) : "memory");
}
__device__ __forceinline__ void wait_async_zero() {
    asm volatile("s_wait_asynccnt 0x0" ::: "memory");
}

// ---------------------------------------------------------------------------
// K0: transpose caps weights  caps_w[cout(256)][cin(256)][9][9] f32
//     -> wT[kk(81)][cout(256)][cin(256)] f16  (cin contiguous)
// ---------------------------------------------------------------------------
__global__ void k_wt(const float* __restrict__ w, f16* __restrict__ wT) {
    int idx = blockIdx.x * 256 + threadIdx.x;        // 81*256*256 = 5308416
    if (idx >= 81 * 256 * 256) return;
    int ci = idx & 255;
    int c  = (idx >> 8) & 255;
    int kk = idx >> 16;
    wT[idx] = (f16)w[(c * 256 + ci) * 81 + kk];
}

// ---------------------------------------------------------------------------
// K1: conv1 + bias + relu, written NHWC f16: x[b][28][28] -> hT[b][y][x][c]
//     image patch staged into LDS with async b128 copies
// ---------------------------------------------------------------------------
__global__ void k_conv1(const float* __restrict__ x, const float* __restrict__ w,
                        const float* __restrict__ bias, f16* __restrict__ hT) {
    __shared__ __align__(16) float patch[9 * 28];    // rows y..y+8, 252 floats
    int y = blockIdx.x;            // 0..19
    int b = blockIdx.y;            // 0..179
    int c = threadIdx.x;           // 0..255
    const float* src = x + b * 784 + y * 28;         // contiguous 1008 bytes
    if (c < 63) async_ld16(&patch[c * 4], src + c * 4);
    wait_async_zero();
    __syncthreads();

    const float* wc = w + c * 81;
    float bv = bias[c];
    float acc[20];
#pragma unroll
    for (int i = 0; i < 20; ++i) acc[i] = bv;

    for (int ky = 0; ky < 9; ++ky) {
        for (int kx = 0; kx < 9; ++kx) {
            float wv = wc[ky * 9 + kx];
            const float* row = &patch[ky * 28 + kx];
#pragma unroll
            for (int xo = 0; xo < 20; ++xo) acc[xo] += wv * row[xo];
        }
    }
    f16* out = hT + (size_t)((b * 20 + y) * 20) * 256 + c;
    for (int xo = 0; xo < 20; ++xo) {
        float r = acc[xo] > 0.f ? acc[xo] : 0.f;
        out[xo * 256] = (f16)r;
    }
}

// ---------------------------------------------------------------------------
// K2: caps conv as implicit GEMM per batch, WMMA f32<-f16 16x16x32.
//     C[p(48 rows, 36 valid), cout(256)] per batch.
//     One wave = one 16-col N tile x all three 16-row M tiles (3 accumulators
//     sharing each B fragment).  Invalid rows are CLAMPED (not branched) so
//     loads are unconditional; fragments are double-buffered so loads for
//     iteration n+1 overlap the 3 WMMAs of iteration n.
// ---------------------------------------------------------------------------
__global__ void k_caps(const f16* __restrict__ hT, const f16* __restrict__ wT,
                       const float* __restrict__ caps_b, float* __restrict__ u) {
    int wave = threadIdx.x >> 5;
    int lane = threadIdx.x & 31;
    int ntile = blockIdx.x * 4 + wave;     // 0..15
    int b = blockIdx.y;
    int hi = lane >> 4;                    // lane group 0/1
    int ln = lane & 15;

    // A row base pointers for the three M tiles; rows >=36 clamped to 35
    // (their results are computed but never stored)
    const f16* pA[3];
#pragma unroll
    for (int mt = 0; mt < 3; ++mt) {
        int p = mt * 16 + ln;
        if (p > 35) p = 35;
        int py = p / 6, px = p % 6;
        pA[mt] = hT + (size_t)(((b * 20 + 2 * py) * 20) + 2 * px) * 256 + hi * 16;
    }
    int cg = ntile * 16 + ln;              // output channel 0..255
    const f16* pB0 = wT + (size_t)cg * 256 + hi * 16;

    v8f acc0, acc1, acc2;
#pragma unroll
    for (int e = 0; e < 8; ++e) { acc0[e] = 0.f; acc1[e] = 0.f; acc2[e] = 0.f; }

    auto loadA = [&](const f16* base, int off) -> v16h {
        v8h lo = *(const v8h*)(base + off);          // K klo..klo+7
        v8h h8 = *(const v8h*)(base + off + 8);      // K klo+8..klo+15
        v16h r;
#pragma unroll
        for (int e = 0; e < 8; ++e) { r[e] = lo[e]; r[e + 8] = h8[e]; }
        return r;
    };

    // prologue: fragments for (kk=0, cc=0)
    v16h a0c = loadA(pA[0], 0);
    v16h a1c = loadA(pA[1], 0);
    v16h a2c = loadA(pA[2], 0);
    v16h bc  = *(const v16h*)pB0;

    for (int kk = 0; kk < 81; ++kk) {
        if (kk < 80)
            __builtin_prefetch(pB0 + (size_t)(kk + 1) * 65536, 0, 1);
#pragma unroll
        for (int cc = 0; cc < 8; ++cc) {
            int nit = kk * 8 + cc + 1;               // next (kk,cc), flattened
            if (nit > 647) nit = 647;                // redundant reload at end
            int kkn = nit >> 3, ccn = nit & 7;
            int kyn = kkn / 9, kxn = kkn - kyn * 9;
            int aoff = (kyn * 20 + kxn) * 256 + ccn * 32;
            size_t boff = (size_t)kkn * 65536 + ccn * 32;
            v16h a0n = loadA(pA[0], aoff);
            v16h a1n = loadA(pA[1], aoff);
            v16h a2n = loadA(pA[2], aoff);
            v16h bn  = *(const v16h*)(pB0 + boff);
            acc0 = __builtin_amdgcn_wmma_f32_16x16x32_f16(
                     false, a0c, false, bc, (short)0, acc0, false, false);
            acc1 = __builtin_amdgcn_wmma_f32_16x16x32_f16(
                     false, a1c, false, bc, (short)0, acc1, false, false);
            acc2 = __builtin_amdgcn_wmma_f32_16x16x32_f16(
                     false, a2c, false, bc, (short)0, acc2, false, false);
            a0c = a0n; a1c = a1n; a2c = a2n; bc = bn;
        }
    }

    // D layout: VGPR v -> M = v + hi*8, N = ln.  u index: cap=cg/32, i=(cg%32)*36+p
    float bv = caps_b[cg];
    int cap = cg >> 5;
    int j   = cg & 31;
    float* ub = u + (size_t)(b * 8 + cap) * 1152 + j * 36;
    v8f accs[3] = {acc0, acc1, acc2};
#pragma unroll
    for (int mt = 0; mt < 3; ++mt) {
#pragma unroll
        for (int v = 0; v < 8; ++v) {
            int pp = mt * 16 + v + hi * 8;
            if (pp < 36) ub[pp] = accs[mt][v] + bv;
        }
    }
}

// ---------------------------------------------------------------------------
// K3: squash u over axis 2 (1152) per (b,cap): u *= mod/((1+mod)*mod_sq)
// ---------------------------------------------------------------------------
__global__ void k_squash_u(float* __restrict__ u) {
    __shared__ float red[128];
    float* up = u + (size_t)blockIdx.x * 1152;   // blockIdx.x = b*8+cap
    int t = threadIdx.x;                          // 0..127, 9 elems each
    float vals[9];
    float ss = 0.f;
#pragma unroll
    for (int j = 0; j < 9; ++j) { float v = up[t * 9 + j]; vals[j] = v; ss += v * v; }
    red[t] = ss; __syncthreads();
    for (int s = 64; s > 0; s >>= 1) { if (t < s) red[t] += red[t + s]; __syncthreads(); }
    float msq = red[0];
    float mod = sqrtf(msq);
    float f = mod / ((1.f + mod) * msq);
#pragma unroll
    for (int j = 0; j < 9; ++j) up[t * 9 + j] = vals[j] * f;
}

__global__ void k_zero(float* __restrict__ p, int n) {
    int i = blockIdx.x * 256 + threadIdx.x;
    if (i < n) p[i] = 0.f;
}

// ---------------------------------------------------------------------------
// K_softmax: c_ij = softmax(b_ij, axis=1) over the 10 out-caps, per i
// ---------------------------------------------------------------------------
__global__ void k_softmax(const float* __restrict__ bij, float* __restrict__ cij) {
    int i = blockIdx.x * 128 + threadIdx.x;
    if (i >= 1152) return;
    float v[10]; float mx = -1e30f;
#pragma unroll
    for (int o = 0; o < 10; ++o) { v[o] = bij[i * 10 + o]; mx = fmaxf(mx, v[o]); }
    float s = 0.f;
#pragma unroll
    for (int o = 0; o < 10; ++o) { v[o] = expf(v[o] - mx); s += v[o]; }
    float inv = 1.f / s;
#pragma unroll
    for (int o = 0; o < 10; ++o) cij[i * 10 + o] = v[o] * inv;
}

// ---------------------------------------------------------------------------
// K4: s[b,o,d] = sum_i c[i,o] * (sum_k W[i,o,d,k] * u[b,k,i])
// ---------------------------------------------------------------------------
__global__ void k_s(const float* __restrict__ cij, const float* __restrict__ W,
                    const float* __restrict__ u, float* __restrict__ s) {
    __shared__ float red[256];
    int o = blockIdx.x, b = blockIdx.y;
    int t = threadIdx.x;
    int d = t & 15, chunk = t >> 4;               // 16 chunks x 16 d
    const float* ub = u + (size_t)b * 8 * 1152;
    float acc = 0.f;
    for (int i = chunk; i < 1152; i += 16) {
        const float* wp = W + (size_t)((i * 10 + o) * 16 + d) * 8;
        float uh = 0.f;
#pragma unroll
        for (int k = 0; k < 8; ++k) uh += wp[k] * ub[k * 1152 + i];
        acc += cij[i * 10 + o] * uh;
    }
    red[t] = acc; __syncthreads();
    for (int st = 128; st >= 16; st >>= 1) { if (t < st) red[t] += red[t + st]; __syncthreads(); }
    if (t < 16) s[(b * 10 + o) * 16 + t] = red[t];
}

// ---------------------------------------------------------------------------
// K5: v = squash(s, axis=1) (over the 10 out caps, per (b,d)); writes d_out
// ---------------------------------------------------------------------------
__global__ void k_v(const float* __restrict__ s, float* __restrict__ v) {
    __shared__ float sq[160];
    __shared__ float msq[16];
    int b = blockIdx.x, t = threadIdx.x;          // t = o*16 + d, 160 threads
    int d = t & 15;
    float sv = s[b * 160 + t];
    sq[t] = sv * sv; __syncthreads();
    if (t < 16) {
        float a = 0.f;
        for (int oo = 0; oo < 10; ++oo) a += sq[oo * 16 + t];
        msq[t] = a;
    }
    __syncthreads();
    float m2 = msq[d];
    float mod = sqrtf(m2);
    v[b * 160 + t] = sv * mod / ((1.f + mod) * m2);
}

// ---------------------------------------------------------------------------
// K6: b_ij[i,o] += (1/B) * sum_b sum_d u_hat[b,i,o,d] * v[b,o,d]
// ---------------------------------------------------------------------------
__global__ void k_bupd(const float* __restrict__ W, const float* __restrict__ u,
                       const float* __restrict__ v, float* __restrict__ bij) {
    __shared__ float red[192];
    int o = blockIdx.x, i = blockIdx.y;
    int t = threadIdx.x;                           // 192 threads, 180 active
    float acc = 0.f;
    if (t < NBATCH) {
        const float* ub = u + (size_t)t * 8 * 1152 + i;
        const float* vb = v + t * 160 + o * 16;
        const float* wp = W + (size_t)(i * 10 + o) * 16 * 8;
        float uvec[8];
#pragma unroll
        for (int k = 0; k < 8; ++k) uvec[k] = ub[k * 1152];
#pragma unroll
        for (int d = 0; d < 16; ++d) {
            float uh = 0.f;
#pragma unroll
            for (int k = 0; k < 8; ++k) uh += wp[d * 8 + k] * uvec[k];
            acc += uh * vb[d];
        }
    }
    red[t] = acc; __syncthreads();
    for (int st = 96; st > 2; st >>= 1) { if (t < st) red[t] += red[t + st]; __syncthreads(); }
    if (t == 0) bij[i * 10 + o] += (red[0] + red[1] + red[2]) * (1.0f / NBATCH);
}

// ---------------------------------------------------------------------------
extern "C" void kernel_launch(void* const* d_in, const int* in_sizes, int n_in,
                              void* d_out, int out_size, void* d_ws, size_t ws_size,
                              hipStream_t stream) {
    const float* x      = (const float*)d_in[0];   // [180,1,28,28]
    const float* conv1w = (const float*)d_in[1];   // [256,1,9,9]
    const float* conv1b = (const float*)d_in[2];   // [256]
    const float* capsw  = (const float*)d_in[3];   // [8,32,256,9,9] == [256,256,9,9]
    const float* capsb  = (const float*)d_in[4];   // [8,32] == [256]
    const float* Wroute = (const float*)d_in[5];   // [1,1152,10,16,8]
    float* vout = (float*)d_out;                   // [180,10,16,1] -> also the v buffer

    char* ws = (char*)d_ws;
    size_t off = 0;
    auto alloc = [&](size_t bytes) {
        void* p = ws + off;
        off = (off + bytes + 255) & ~(size_t)255;
        return p;
    };
    f16*   wT  = (f16*)alloc((size_t)81 * 256 * 256 * sizeof(f16));         // 10.6 MB
    f16*   hT  = (f16*)alloc((size_t)NBATCH * 20 * 20 * 256 * sizeof(f16)); // 36.9 MB
    float* u   = (float*)alloc((size_t)NBATCH * 8 * 1152 * sizeof(float));  // 6.6 MB
    float* bij = (float*)alloc(1152 * 10 * sizeof(float));
    float* cij = (float*)alloc(1152 * 10 * sizeof(float));
    float* s   = (float*)alloc(NBATCH * 160 * sizeof(float));

    k_wt      <<<(81 * 256 * 256 + 255) / 256, 256, 0, stream>>>(capsw, wT);
    k_conv1   <<<dim3(20, NBATCH), 256, 0, stream>>>(x, conv1w, conv1b, hT);
    k_caps    <<<dim3(4, NBATCH), 128, 0, stream>>>(hT, wT, capsb, u);
    k_squash_u<<<NBATCH * 8, 128, 0, stream>>>(u);
    k_zero    <<<(11520 + 255) / 256, 256, 0, stream>>>(bij, 11520);

    for (int it = 0; it < 3; ++it) {
        k_softmax<<<9, 128, 0, stream>>>(bij, cij);
        k_s      <<<dim3(10, NBATCH), 256, 0, stream>>>(cij, Wroute, u, s);
        k_v      <<<NBATCH, 160, 0, stream>>>(s, vout);
        if (it < 2)
            k_bupd<<<dim3(10, 1152), 192, 0, stream>>>(Wroute, u, vout, bij);
    }
}
